// DTAModel_60017872994809
// MI455X (gfx1250) — matrix operand order repaired and checked by reading, not compiled, not run on hardware
//
#include <hip/hip_runtime.h>
#include <hip/hip_bf16.h>
#include <math.h>

// ---------------- problem constants ----------------
#define B_   16
#define ND_  48
#define ED_  192
#define NPn_ 256
#define EPe_ 4096

#define USE_ASYNC_LDS 1   // gfx1250 async global->LDS staging (verified to assemble)

typedef __attribute__((ext_vector_type(16))) __bf16 v16bf;
typedef __attribute__((ext_vector_type(8)))  float  v8f;
typedef __attribute__((address_space(3))) float lds_f32;

__device__ __forceinline__ float sigf(float x) { return 1.f / (1.f + expf(-x)); }

// =====================================================================
// WMMA GEMM:  C[m, n] = act( sum_k A[m, k] * W[k, n] + bias[n] )
//   A element addr : A + z*bA + m*lda + k*kst
//   W element addr : W + k*ldw + n
//   C element addr : C + z*bC + m*ldc + n*cst
// Block = 128 threads (4 waves). Each wave computes a 32x32 output tile
// (2 A-frags x 2 B-frags -> 4 v_wmma per 32-deep K step, f32 accumulate).
// All 4 waves share the block's 32-wide N tile; the W[k0:k0+32, 32] tile
// is staged once per block into LDS via async global->LDS DMA with
// strength-reduced (incremental) addressing. Full K chunks are fully
// branchless; only the K tail uses selects.
// =====================================================================
__global__ void k_gemm(const float* __restrict__ A, long lda, long kst, long bA,
                       const float* __restrict__ W, long ldw,
                       const float* __restrict__ bias,
                       float* __restrict__ C, long ldc, long cst, long bC,
                       int M, int N, int K, int act)
{
    __shared__ float lds_w[32][32];     // [col-in-tile][k-in-chunk]

    int tid  = threadIdx.x;
    int lane = tid & 31;
    int wv   = tid >> 5;
    int tnBase = blockIdx.x * 32;
    int tmBase = (blockIdx.y * 4 + wv) * 32;
    long z = blockIdx.z;

    const float* Az = A + z * bA;
    float*       Cz = C + z * bC;

    // clamped A row pointers for the two 16-row sub-tiles
    int m0 = tmBase + (lane & 15);        if (m0 >= M) m0 = M - 1;
    int m1 = tmBase + 16 + (lane & 15);   if (m1 >= M) m1 = M - 1;
    const float* Ar0 = Az + (long)m0 * lda;
    const float* Ar1 = Az + (long)m1 * lda;

    int aoff = (lane >> 4) << 3;   // +8  K offset for lanes 16..31 (A frag)
    int boff = (lane >> 4) << 4;   // (unused) B frag uses same K order as A frag
    int cc0  = lane & 15;          // B frag column within tile

    // ---- precomputed staging coordinates (element e = tid + j*128) ----
    const float* gps[8];           // global source pointers (advance by 32*ldw)
    unsigned     loffs[8];         // LDS byte offsets (loop-invariant)
    int          kks[8], ccs[8];   // k-in-chunk / col-in-tile per slot
#pragma unroll
    for (int j = 0; j < 8; ++j) {
        int e  = tid + j * 128;
        int kk = e >> 5, cc = e & 31;
        int ng = tnBase + cc;
        int ngc = (ng < N) ? ng : N - 1;          // pad cols unused downstream
        kks[j] = kk; ccs[j] = cc;
        gps[j] = W + (long)kk * ldw + ngc;
        loffs[j] = (unsigned)(size_t)(lds_f32*)&lds_w[cc][kk];
    }
    const long wstep = 32 * ldw;

    v8f acc00, acc01, acc10, acc11;
#pragma unroll
    for (int g = 0; g < 8; ++g) { acc00[g] = 0.f; acc01[g] = 0.f; acc10[g] = 0.f; acc11[g] = 0.f; }

    for (int k0 = 0; k0 < K; k0 += 32) {
        bool full = (k0 + 32 <= K);     // block-uniform
        __syncthreads();                // previous chunk's LDS reads complete
        if (full) {
            // ---- branchless staging: 8 async DMA copies per thread ----
#pragma unroll
            for (int j = 0; j < 8; ++j) {
#if USE_ASYNC_LDS
                asm volatile("global_load_async_to_lds_b32 %0, %1, off"
                             :: "v"(loffs[j]), "v"(gps[j]) : "memory");
#else
                lds_w[ccs[j]][kks[j]] = *gps[j];
#endif
            }
#if USE_ASYNC_LDS
            asm volatile("s_wait_asynccnt 0" ::: "memory");
#endif
        } else {
            // ---- K tail: zero the pad rows, DMA the valid ones ----
#pragma unroll
            for (int j = 0; j < 8; ++j) {
#if USE_ASYNC_LDS
                if (k0 + kks[j] < K) {
                    asm volatile("global_load_async_to_lds_b32 %0, %1, off"
                                 :: "v"(loffs[j]), "v"(gps[j]) : "memory");
                } else {
                    lds_w[ccs[j]][kks[j]] = 0.f;
                }
#else
                float wval = (k0 + kks[j] < K) ? *gps[j] : 0.f;
                lds_w[ccs[j]][kks[j]] = wval;
#endif
            }
#if USE_ASYNC_LDS
            asm volatile("s_wait_asynccnt 0" ::: "memory");
#endif
        }
#pragma unroll
        for (int j = 0; j < 8; ++j) gps[j] += wstep;   // incremental addressing
        __syncthreads();

        if (k0 + 32 < K) __builtin_prefetch(Ar0 + (long)(k0 + 32) * kst, 0, 3);

        v16bf a0, a1, b0, b1;
        if (full) {
            if (kst == 1) {
                // contiguous runs: let the vectorizer merge into wide loads
                const float* p0 = Ar0 + k0 + aoff;
                const float* p1 = Ar1 + k0 + aoff;
#pragma unroll
                for (int i = 0; i < 8; ++i) {
                    a0[i]     = (__bf16)p0[i];
                    a0[i + 8] = (__bf16)p0[16 + i];
                    a1[i]     = (__bf16)p1[i];
                    a1[i + 8] = (__bf16)p1[16 + i];
                }
            } else {
#pragma unroll
                for (int i = 0; i < 16; ++i) {
                    int ka = k0 + (i & 7) + ((i >> 3) << 4) + aoff;
                    a0[i] = (__bf16)Ar0[(long)ka * kst];
                    a1[i] = (__bf16)Ar1[(long)ka * kst];
                }
            }
        } else {                                   // K tail: select-zeroed
#pragma unroll
            for (int i = 0; i < 16; ++i) {
                int ka  = k0 + (i & 7) + ((i >> 3) << 4) + aoff;
                int kac = (ka < K) ? ka : K - 1;
                float f0 = Ar0[(long)kac * kst];
                float f1 = Ar1[(long)kac * kst];
                a0[i] = (__bf16)((ka < K) ? f0 : 0.f);
                a1[i] = (__bf16)((ka < K) ? f1 : 0.f);
            }
        }
#pragma unroll
        for (int i = 0; i < 16; ++i) {
            int kk = aoff + (i & 7) + ((i >> 3) << 4);
            b0[i] = (__bf16)lds_w[cc0][kk];
            b1[i] = (__bf16)lds_w[cc0 + 16][kk];
        }
        acc00 = __builtin_amdgcn_wmma_f32_16x16x32_bf16(false, a0, false, b0, (short)0, acc00, false, false);
        acc01 = __builtin_amdgcn_wmma_f32_16x16x32_bf16(false, a0, false, b1, (short)0, acc01, false, false);
        acc10 = __builtin_amdgcn_wmma_f32_16x16x32_bf16(false, a1, false, b0, (short)0, acc10, false, false);
        acc11 = __builtin_amdgcn_wmma_f32_16x16x32_bf16(false, a1, false, b1, (short)0, acc11, false, false);
    }

    // ---- store 32x32 tile (guarded) ----
#pragma unroll
    for (int mt = 0; mt < 2; ++mt) {
#pragma unroll
        for (int nt = 0; nt < 2; ++nt) {
            v8f acc = (mt == 0) ? (nt == 0 ? acc00 : acc01) : (nt == 0 ? acc10 : acc11);
            int col = tnBase + nt * 16 + (lane & 15);
            bool ok = (col < N);
            float bb = (bias && ok) ? bias[col] : 0.f;
            int rbase = tmBase + mt * 16 + ((lane >> 4) << 3);
#pragma unroll
            for (int g = 0; g < 8; ++g) {
                int r = rbase + g;
                if (r < M && ok) {
                    float vv = acc[g] + bb;
                    if (act == 1) vv = fmaxf(vv, 0.f);
                    Cz[(long)r * ldc + (long)col * cst] = vv;
                }
            }
        }
    }
}

// =====================================================================
// Elementwise / graph kernels
// =====================================================================
__global__ void k_zero(float* p, long n)
{
    long t = (long)blockIdx.x * blockDim.x + threadIdx.x;
    if (t < n) p[t] = 0.f;
}

__global__ void k_copy_cols(float* dst, long ldd, long dstoff,
                            const float* src, long lds, long srcoff,
                            int rows, int w)
{
    long t = (long)blockIdx.x * blockDim.x + threadIdx.x;
    if (t >= (long)rows * w) return;
    int r = (int)(t / w), c = (int)(t % w);
    dst[(long)r * ldd + dstoff + c] = src[(long)r * lds + srcoff + c];
}

__global__ void k_gather_cols(float* dst, long ldd, long dstoff,
                              const float* src, long lds,
                              const int* __restrict__ idx, int rows, int w)
{
    long t = (long)blockIdx.x * blockDim.x + threadIdx.x;
    if (t >= (long)rows * w) return;
    int r = (int)(t / w), c = (int)(t % w);
    dst[(long)r * ldd + dstoff + c] = src[(long)idx[r] * lds + c];
}

__global__ void k_vnorms(const float* __restrict__ v, int H,
                         float* dst, long ldd, long off, int rows)
{
    long t = (long)blockIdx.x * blockDim.x + threadIdx.x;
    if (t >= (long)rows * H) return;
    int r = (int)(t / H), h = (int)(t % H);
    const float* p = v + ((long)r * H + h) * 3;
    float ss = p[0] * p[0] + p[1] * p[1] + p[2] * p[2];
    dst[(long)r * ldd + off + h] = sqrtf(fmaxf(ss, 1e-8f));
}

__global__ void k_vgate(float* v, int H, long rows)
{
    long t = (long)blockIdx.x * blockDim.x + threadIdx.x;
    if (t >= rows * H) return;
    long r = t / H; int h = (int)(t % H);
    float* p = v + (r * H + h) * 3;
    float ss = p[0] * p[0] + p[1] * p[1] + p[2] * p[2];
    float g = sigf(sqrtf(fmaxf(ss, 1e-8f)));
    p[0] *= g; p[1] *= g; p[2] *= g;
}

__global__ void k_ln(const float* __restrict__ s, const float* __restrict__ dhs,
                     const float* __restrict__ cnt,
                     const float* __restrict__ g, const float* __restrict__ b,
                     float* __restrict__ s_out,
                     const float* __restrict__ v, const float* __restrict__ dhv,
                     float* __restrict__ v_out,
                     int N, int D, int NVc)
{
    int n = blockIdx.x * blockDim.x + threadIdx.x;
    if (n >= N) return;
    float inv = cnt ? 1.f / cnt[n] : 1.f;
    const float* sr = s + (long)n * D;
    const float* dr = dhs ? dhs + (long)n * D : nullptr;
    float mu = 0.f, m2 = 0.f;
    for (int d = 0; d < D; ++d) {
        float x = sr[d] + (dr ? dr[d] * inv : 0.f);
        mu += x; m2 += x * x;
    }
    mu /= D;
    float var = m2 / D - mu * mu;
    float rs = rsqrtf(var + 1e-5f);
    for (int d = 0; d < D; ++d) {
        float x = sr[d] + (dr ? dr[d] * inv : 0.f);
        s_out[(long)n * D + d] = (x - mu) * rs * g[d] + b[d];
    }
    if (v) {
        const float* vr = v + (long)n * NVc * 3;
        const float* dv = dhv ? dhv + (long)n * NVc * 3 : nullptr;
        float acc = 0.f;
        for (int h = 0; h < NVc; ++h) {
            float ss = 0.f;
            for (int c = 0; c < 3; ++c) {
                float x = vr[h * 3 + c] + (dv ? dv[h * 3 + c] * inv : 0.f);
                ss += x * x;
            }
            acc += fmaxf(ss, 1e-8f);
        }
        float sc = rsqrtf(acc / NVc);
        for (int h = 0; h < NVc; ++h)
            for (int c = 0; c < 3; ++c) {
                float x = vr[h * 3 + c] + (dv ? dv[h * 3 + c] * inv : 0.f);
                v_out[(long)n * NVc * 3 + h * 3 + c] = x * sc;
            }
    }
}

__global__ void k_count(const int* __restrict__ dst, float* cnt, int E)
{
    int e = blockIdx.x * blockDim.x + threadIdx.x;
    if (e < E) atomicAdd(&cnt[dst[e]], 1.f);
}

__global__ void k_clamp1(float* cnt, int N)
{
    int n = blockIdx.x * blockDim.x + threadIdx.x;
    if (n < N) cnt[n] = fmaxf(cnt[n], 1.f);
}

__global__ void k_scatter_add(const float* __restrict__ src, const int* __restrict__ dst,
                              float* out, int E, int W)
{
    long t = (long)blockIdx.x * blockDim.x + threadIdx.x;
    if (t >= (long)E * W) return;
    int e = (int)(t / W), c = (int)(t % W);
    atomicAdd(&out[(long)dst[e] * W + c], src[(long)e * W + c]);
}

__global__ void k_softmax_rows(float* x, int rows, int L)
{
    int r = blockIdx.x * blockDim.x + threadIdx.x;
    if (r >= rows) return;
    float* p = x + (long)r * L;
    float m = -1e30f;
    for (int l = 0; l < L; ++l) m = fmaxf(m, p[l]);
    float s = 0.f;
    for (int l = 0; l < L; ++l) s += expf(p[l] - m);
    float inv = 1.f / s;
    for (int l = 0; l < L; ++l) p[l] = expf(p[l] - m) * inv;
}

__global__ void k_colsum(const float* __restrict__ att, float* S, int Bn, int L, int cn)
{
    long t = (long)blockIdx.x * blockDim.x + threadIdx.x;
    if (t >= (long)Bn * L) return;
    int b = (int)(t / L), l = (int)(t % L);
    float s = 0.f;
    for (int n = 0; n < cn; ++n) s += att[((long)b * cn + n) * L + l];
    S[(long)b * L + l] = s;
}

__global__ void k_gmean(const float* __restrict__ A, const float* __restrict__ S,
                        float* g, int Bn, int L, int cm)
{
    long t = (long)blockIdx.x * blockDim.x + threadIdx.x;
    if (t >= (long)Bn * cm) return;
    int b = (int)(t / cm), m = (int)(t % cm);
    float s = 0.f;
    for (int l = 0; l < L; ++l) s += A[((long)b * cm + m) * L + l] * S[(long)b * L + l];
    g[(long)b * cm + m] = s * (1.f / 128.f);
}

__global__ void k_att_sig(const float* __restrict__ gd, const float* __restrict__ tt,
                          float* att, int Bn, int L, int na)
{
    long t = (long)blockIdx.x * blockDim.x + threadIdx.x;
    if (t >= (long)Bn * L) return;
    int b = (int)(t / L), l = (int)(t % L);
    float s = 0.f;
    for (int a = 0; a < na; ++a) s += gd[(long)b * na + a] * tt[((long)b * na + a) * L + l];
    att[(long)b * L + l] = sigf(s);
}

__global__ void k_mut_out(const float* __restrict__ enc, const float* __restrict__ att,
                          float* out, int Bn, int L)
{
    long t = (long)blockIdx.x * blockDim.x + threadIdx.x;
    if (t >= (long)Bn * 128 * L) return;
    int b = (int)(t / (128L * L));
    int rem = (int)(t % (128L * L));
    int c = rem / L, l = rem % L;
    float x = enc[((long)b * L + l) * 128 + c];
    float a = att[(long)b * L + l];
    out[((long)b * 128 + c) * L + l] = fmaxf(x * (1.f + a), 0.f);
}

__global__ void k_rowmax(const float* __restrict__ src, int L,
                         float* dst, int ldd, int off, int Bn)
{
    long t = (long)blockIdx.x * blockDim.x + threadIdx.x;
    if (t >= (long)Bn * 128) return;
    int b = (int)(t / 128), c = (int)(t % 128);
    const float* p = src + ((long)b * 128 + c) * L;
    float m = -1e30f;
    for (int l = 0; l < L; ++l) m = fmaxf(m, p[l]);
    dst[(long)b * ldd + off + c] = m;
}

// =====================================================================
// Host-side orchestration
// =====================================================================
struct GvpP { const float *bs, *wh, *ws, *wv; };
struct LnP  { const float *b, *g; };
struct LayerP { GvpP conv[3]; GvpP ff[2]; LnP ln0, ln1; };
struct EncP { GvpP W_e, W_out, W_v; LayerP L[3]; LnP ln_e, ln_out, ln_v; };
struct MaP  { const float *Sb, *Sw, *Tb, *Tw, *Ab, *Aw, *Bb, *Bw, *lb, *lw; };
struct MlpP { const float *b1, *b2, *b3, *w1, *w2, *w3; };
struct Ws   { float *S,*V,*ES,*EV,*CNT,*MS,*MV,*VH,*SA,*SB,*DHS,*DHV,*NS1,*NS2,*NV1; };

static void loadGvp(GvpP& g, void* const* d, int& i, bool wv)
{
    g.bs = (const float*)d[i++]; g.wh = (const float*)d[i++];
    g.ws = (const float*)d[i++]; g.wv = wv ? (const float*)d[i++] : nullptr;
}
static void loadLn(LnP& l, void* const* d, int& i)
{
    l.b = (const float*)d[i++]; l.g = (const float*)d[i++];
}
static void loadEnc(EncP& e, void* const* d, int& i)
{
    loadGvp(e.W_e, d, i, true);
    loadGvp(e.W_out, d, i, false);
    loadGvp(e.W_v, d, i, true);
    for (int k = 0; k < 3; ++k) {
        for (int c = 0; c < 3; ++c) loadGvp(e.L[k].conv[c], d, i, true);
        loadGvp(e.L[k].ff[0], d, i, true);
        loadGvp(e.L[k].ff[1], d, i, true);
        loadLn(e.L[k].ln0, d, i);
        loadLn(e.L[k].ln1, d, i);
    }
    loadLn(e.ln_e, d, i); loadLn(e.ln_out, d, i); loadLn(e.ln_v, d, i);
}
static void loadMa(MaP& m, void* const* d, int& i)
{
    m.Sb = (const float*)d[i++]; m.Sw = (const float*)d[i++];
    m.Tb = (const float*)d[i++]; m.Tw = (const float*)d[i++];
    m.Ab = (const float*)d[i++]; m.Aw = (const float*)d[i++];
    m.Bb = (const float*)d[i++]; m.Bw = (const float*)d[i++];
    m.lb = (const float*)d[i++]; m.lw = (const float*)d[i++];
}
static void loadMlp(MlpP& m, void* const* d, int& i)
{
    m.b1 = (const float*)d[i++]; m.b2 = (const float*)d[i++]; m.b3 = (const float*)d[i++];
    m.w1 = (const float*)d[i++]; m.w2 = (const float*)d[i++]; m.w3 = (const float*)d[i++];
}

static inline unsigned g1(long n, int bs) { return (unsigned)((n + bs - 1) / bs); }

static void gemm(hipStream_t st, const float* A, long lda, long kst, long bA,
                 const float* W, long ldw, const float* bias,
                 float* C, long ldc, long cst, long bC,
                 int M, int N, int K, int act, int nb)
{
    dim3 grid((N + 31) / 32, (M + 127) / 128, nb);
    k_gemm<<<grid, dim3(128, 1, 1), 0, st>>>(A, lda, kst, bA, W, ldw, bias,
                                             C, ldc, cst, bC, M, N, K, act);
}

static void run_encoder(hipStream_t st, const EncP& P, int Nn, int E,
                        int si, int vi, int se,
                        const float* ns, const float* nv,
                        const float* esi, const float* evi,
                        const int* eidx, const Ws& w, float* out)
{
    const int* srcI = eidx;
    const int* dstI = eidx + E;
    const int TB = 256;

    // ---- front: s,v = GVP(W_v, ln_v(node)) ----
    k_ln<<<g1(Nn, 128), 128, 0, st>>>(ns, nullptr, nullptr, P.ln_v.g, P.ln_v.b, w.NS1,
                                      nv, nullptr, w.NV1, Nn, si, vi);
    gemm(st, w.NV1, (long)vi * 3, 3, 1, P.W_v.wh, 64, nullptr, w.VH, 192, 3, 1, Nn, 64, vi, 0, 3);
    k_copy_cols<<<g1((long)Nn * si, TB), TB, 0, st>>>(w.NS2, si + 64, 0, w.NS1, si, 0, Nn, si);
    k_vnorms<<<g1((long)Nn * 64, TB), TB, 0, st>>>(w.VH, 64, w.NS2, si + 64, si, Nn);
    gemm(st, w.NS2, si + 64, 1, 0, P.W_v.ws, 128, P.W_v.bs, w.S, 128, 1, 0, Nn, 128, si + 64, 0, 1);
    gemm(st, w.VH, 192, 3, 1, P.W_v.wv, 64, nullptr, w.V, 192, 3, 1, Nn, 64, 64, 0, 3);

    // ---- front edges: es,ev = GVP(W_e, ln_e(edge)) ----
    k_ln<<<g1(E, 128), 128, 0, st>>>(esi, nullptr, nullptr, P.ln_e.g, P.ln_e.b, w.NS1,
                                     evi, nullptr, w.NV1, E, se, 1);
    gemm(st, w.NV1, 3, 3, 1, P.W_e.wh, 1, nullptr, w.VH, 3, 3, 1, E, 1, 1, 0, 3);
    k_copy_cols<<<g1((long)E * se, TB), TB, 0, st>>>(w.MS, se + 1, 0, w.NS1, se, 0, E, se);
    k_vnorms<<<g1((long)E, TB), TB, 0, st>>>(w.VH, 1, w.MS, se + 1, se, E);
    gemm(st, w.MS, se + 1, 1, 0, P.W_e.ws, 32, P.W_e.bs, w.ES, 32, 1, 0, E, 32, se + 1, 0, 1);
    gemm(st, w.VH, 3, 3, 1, P.W_e.wv, 1, nullptr, w.EV, 3, 3, 1, E, 1, 1, 0, 3);

    // ---- in-degree counts ----
    k_zero<<<g1(Nn, TB), TB, 0, st>>>(w.CNT, Nn);
    k_count<<<g1(E, TB), TB, 0, st>>>(dstI, w.CNT, E);
    k_clamp1<<<g1(Nn, TB), TB, 0, st>>>(w.CNT, Nn);

    for (int l = 0; l < 3; ++l) {
        const LayerP& L = P.L[l];
        // gather messages: ms = [s[src] | es | s[dst]] (417-wide incl norms), mv
        k_gather_cols<<<g1((long)E * 128, TB), TB, 0, st>>>(w.MS, 417, 0, w.S, 128, srcI, E, 128);
        k_copy_cols<<<g1((long)E * 32, TB), TB, 0, st>>>(w.MS, 417, 128, w.ES, 32, 0, E, 32);
        k_gather_cols<<<g1((long)E * 128, TB), TB, 0, st>>>(w.MS, 417, 160, w.S, 128, dstI, E, 128);
        k_gather_cols<<<g1((long)E * 192, TB), TB, 0, st>>>(w.MV, 387, 0, w.V, 192, srcI, E, 192);
        k_copy_cols<<<g1((long)E * 3, TB), TB, 0, st>>>(w.MV, 387, 192, w.EV, 3, 0, E, 3);
        k_gather_cols<<<g1((long)E * 192, TB), TB, 0, st>>>(w.MV, 387, 195, w.V, 192, dstI, E, 192);
        // conv[0]
        gemm(st, w.MV, 387, 3, 1, L.conv[0].wh, 129, nullptr, w.VH, 387, 3, 1, E, 129, 129, 0, 3);
        k_vnorms<<<g1((long)E * 129, TB), TB, 0, st>>>(w.VH, 129, w.MS, 417, 288, E);
        gemm(st, w.MS, 417, 1, 0, L.conv[0].ws, 128, L.conv[0].bs, w.SA, 128, 1, 0, E, 128, 417, 1, 1);
        gemm(st, w.VH, 387, 3, 1, L.conv[0].wv, 64, nullptr, w.MV, 192, 3, 1, E, 64, 129, 0, 3);
        k_vgate<<<g1((long)E * 64, TB), TB, 0, st>>>(w.MV, 64, E);
        // conv[1]
        gemm(st, w.MV, 192, 3, 1, L.conv[1].wh, 64, nullptr, w.VH, 192, 3, 1, E, 64, 64, 0, 3);
        k_copy_cols<<<g1((long)E * 128, TB), TB, 0, st>>>(w.MS, 192, 0, w.SA, 128, 0, E, 128);
        k_vnorms<<<g1((long)E * 64, TB), TB, 0, st>>>(w.VH, 64, w.MS, 192, 128, E);
        gemm(st, w.MS, 192, 1, 0, L.conv[1].ws, 128, L.conv[1].bs, w.SB, 128, 1, 0, E, 128, 192, 1, 1);
        gemm(st, w.VH, 192, 3, 1, L.conv[1].wv, 64, nullptr, w.MV, 192, 3, 1, E, 64, 64, 0, 3);
        k_vgate<<<g1((long)E * 64, TB), TB, 0, st>>>(w.MV, 64, E);
        // conv[2] (no activations)
        gemm(st, w.MV, 192, 3, 1, L.conv[2].wh, 64, nullptr, w.VH, 192, 3, 1, E, 64, 64, 0, 3);
        k_copy_cols<<<g1((long)E * 128, TB), TB, 0, st>>>(w.MS, 192, 0, w.SB, 128, 0, E, 128);
        k_vnorms<<<g1((long)E * 64, TB), TB, 0, st>>>(w.VH, 64, w.MS, 192, 128, E);
        gemm(st, w.MS, 192, 1, 0, L.conv[2].ws, 128, L.conv[2].bs, w.SA, 128, 1, 0, E, 128, 192, 0, 1);
        gemm(st, w.VH, 192, 3, 1, L.conv[2].wv, 64, nullptr, w.MV, 192, 3, 1, E, 64, 64, 0, 3);
        // scatter mean + residual + ln0
        k_zero<<<g1((long)Nn * 128, TB), TB, 0, st>>>(w.DHS, (long)Nn * 128);
        k_zero<<<g1((long)Nn * 192, TB), TB, 0, st>>>(w.DHV, (long)Nn * 192);
        k_scatter_add<<<g1((long)E * 128, TB), TB, 0, st>>>(w.SA, dstI, w.DHS, E, 128);
        k_scatter_add<<<g1((long)E * 192, TB), TB, 0, st>>>(w.MV, dstI, w.DHV, E, 192);
        k_ln<<<g1(Nn, 128), 128, 0, st>>>(w.S, w.DHS, w.CNT, L.ln0.g, L.ln0.b, w.S,
                                          w.V, w.DHV, w.V, Nn, 128, 64);
        // ff[0]
        gemm(st, w.V, 192, 3, 1, L.ff[0].wh, 128, nullptr, w.VH, 384, 3, 1, Nn, 128, 64, 0, 3);
        k_copy_cols<<<g1((long)Nn * 128, TB), TB, 0, st>>>(w.NS2, 256, 0, w.S, 128, 0, Nn, 128);
        k_vnorms<<<g1((long)Nn * 128, TB), TB, 0, st>>>(w.VH, 128, w.NS2, 256, 128, Nn);
        gemm(st, w.NS2, 256, 1, 0, L.ff[0].ws, 512, L.ff[0].bs, w.NS1, 512, 1, 0, Nn, 512, 256, 1, 1);
        gemm(st, w.VH, 384, 3, 1, L.ff[0].wv, 128, nullptr, w.NV1, 384, 3, 1, Nn, 128, 128, 0, 3);
        k_vgate<<<g1((long)Nn * 128, TB), TB, 0, st>>>(w.NV1, 128, Nn);
        // ff[1] (no activations)
        gemm(st, w.NV1, 384, 3, 1, L.ff[1].wh, 128, nullptr, w.VH, 384, 3, 1, Nn, 128, 128, 0, 3);
        k_copy_cols<<<g1((long)Nn * 512, TB), TB, 0, st>>>(w.NS2, 640, 0, w.NS1, 512, 0, Nn, 512);
        k_vnorms<<<g1((long)Nn * 128, TB), TB, 0, st>>>(w.VH, 128, w.NS2, 640, 512, Nn);
        gemm(st, w.NS2, 640, 1, 0, L.ff[1].ws, 128, L.ff[1].bs, w.DHS, 128, 1, 0, Nn, 128, 640, 0, 1);
        gemm(st, w.VH, 384, 3, 1, L.ff[1].wv, 64, nullptr, w.DHV, 192, 3, 1, Nn, 64, 128, 0, 3);
        k_ln<<<g1(Nn, 128), 128, 0, st>>>(w.S, w.DHS, nullptr, L.ln1.g, L.ln1.b, w.S,
                                          w.V, w.DHV, w.V, Nn, 128, 64);
    }

    // ---- out: relu(GVP(W_out, ln_out(s, v))) ----
    k_ln<<<g1(Nn, 128), 128, 0, st>>>(w.S, nullptr, nullptr, P.ln_out.g, P.ln_out.b, w.NS1,
                                      w.V, nullptr, w.NV1, Nn, 128, 64);
    gemm(st, w.NV1, 192, 3, 1, P.W_out.wh, 64, nullptr, w.VH, 192, 3, 1, Nn, 64, 64, 0, 3);
    k_copy_cols<<<g1((long)Nn * 128, TB), TB, 0, st>>>(w.NS2, 192, 0, w.NS1, 128, 0, Nn, 128);
    k_vnorms<<<g1((long)Nn * 64, TB), TB, 0, st>>>(w.VH, 64, w.NS2, 192, 128, Nn);
    gemm(st, w.NS2, 192, 1, 0, P.W_out.ws, 128, P.W_out.bs, out, 128, 1, 0, Nn, 128, 192, 1, 1);
}

static void run_mutual(hipStream_t st, const MaP& M,
                       const float* encS, int Ls, const float* encT, int Lt,
                       float* Abuf, float* Btb, float* Sb_, float* gbuf, float* g2,
                       float* gd, float* Tt, float* attb, float* outT)
{
    const int TB = 256;
    gemm(st, encS, 128, 1, (long)Ls * 128, M.Aw, 128, M.Ab, Abuf, 1, Ls, (long)128 * Ls, Ls, 128, 128, 0, B_);
    gemm(st, encS, 128, 1, (long)Ls * 128, M.Bw, 128, M.Bb, Btb, 1, Ls, (long)128 * Ls, Ls, 128, 128, 0, B_);
    k_softmax_rows<<<g1(B_ * 128, 128), 128, 0, st>>>(Btb, B_ * 128, Ls);
    k_colsum<<<g1((long)B_ * Ls, TB), TB, 0, st>>>(Btb, Sb_, B_, Ls, 128);
    k_gmean<<<g1((long)B_ * 128, TB), TB, 0, st>>>(Abuf, Sb_, gbuf, B_, Ls, 128);
    gemm(st, gbuf, 128, 1, 0, M.lw, 128, M.lb, g2, 128, 1, 0, B_, 128, 128, 0, 1);
    gemm(st, g2, 128, 1, 0, M.Sw, 128, M.Sb, gd, 128, 1, 0, B_, 128, 128, 0, 1);
    gemm(st, encT, 128, 1, (long)Lt * 128, M.Tw, 128, M.Tb, Tt, 1, Lt, (long)128 * Lt, Lt, 128, 128, 0, B_);
    k_att_sig<<<g1((long)B_ * Lt, TB), TB, 0, st>>>(gd, Tt, attb, B_, Lt, 128);
    k_mut_out<<<g1((long)B_ * 128 * Lt, TB), TB, 0, st>>>(encT, attb, outT, B_, Lt);
}

extern "C" void kernel_launch(void* const* d_in, const int* in_sizes, int n_in,
                              void* d_out, int out_size, void* d_ws, size_t ws_size,
                              hipStream_t stream)
{
    (void)in_sizes; (void)n_in; (void)out_size; (void)ws_size;
    const float* dns = (const float*)d_in[0];
    const float* dnv = (const float*)d_in[1];
    const float* des = (const float*)d_in[2];
    const float* dev = (const float*)d_in[3];
    const int*   dei = (const int*)d_in[4];
    const float* pns = (const float*)d_in[5];
    const float* pnv = (const float*)d_in[6];
    const float* pes = (const float*)d_in[7];
    const float* pev = (const float*)d_in[8];
    const int*   pei = (const int*)d_in[9];

    int pi = 10;
    EncP drug, prot; MaP ma_d, ma_p; MlpP mlp;
    loadEnc(drug, d_in, pi);   // 'drug'
    loadMa(ma_d, d_in, pi);    // 'ma_d'
    loadMa(ma_p, d_in, pi);    // 'ma_p'
    loadMlp(mlp, d_in, pi);    // 'mlp'
    loadEnc(prot, d_in, pi);   // 'prot'

    float* base = (float*)d_ws;
    size_t off = 0;
    auto alloc = [&](size_t n) { float* p = base + off; off += (n + 63) & ~(size_t)63; return p; };
    const int NnM = B_ * NPn_, EM = B_ * EPe_;
    Ws w;
    w.S   = alloc((size_t)NnM * 128);
    w.V   = alloc((size_t)NnM * 192);
    w.ES  = alloc((size_t)EM * 32);
    w.EV  = alloc((size_t)EM * 3);
    w.CNT = alloc((size_t)NnM);
    w.MS  = alloc((size_t)EM * 417);
    w.MV  = alloc((size_t)EM * 387);
    w.VH  = alloc((size_t)EM * 387);
    w.SA  = alloc((size_t)EM * 128);
    w.SB  = alloc((size_t)EM * 128);
    w.DHS = alloc((size_t)NnM * 128);
    w.DHV = alloc((size_t)NnM * 192);
    w.NS1 = alloc((size_t)NnM * 640);
    w.NS2 = alloc((size_t)NnM * 640);
    w.NV1 = alloc((size_t)NnM * 384);
    float* ENCD = alloc((size_t)B_ * ND_ * 128);
    float* ENCP = alloc((size_t)B_ * NPn_ * 128);
    float* MU_A = alloc((size_t)B_ * 128 * 256);
    float* MU_B = alloc((size_t)B_ * 128 * 256);
    float* MU_S = alloc((size_t)B_ * 256);
    float* MU_G = alloc((size_t)B_ * 128);
    float* MU_G2= alloc((size_t)B_ * 128);
    float* MU_GD= alloc((size_t)B_ * 128);
    float* MU_T = alloc((size_t)B_ * 128 * 256);
    float* MU_AT= alloc((size_t)B_ * 256);
    float* P2   = alloc((size_t)B_ * 128 * 256);
    float* D2   = alloc((size_t)B_ * 128 * ND_);
    float* XC   = alloc((size_t)B_ * 256);
    float* M1   = alloc((size_t)B_ * 1024);
    float* M2   = alloc((size_t)B_ * 512);

    run_encoder(stream, drug, B_ * ND_, B_ * ED_, 66, 1, 16, dns, dnv, des, dev, dei, w, ENCD);
    run_encoder(stream, prot, B_ * NPn_, B_ * EPe_, 6, 3, 39, pns, pnv, pes, pev, pei, w, ENCP);

    run_mutual(stream, ma_p, ENCD, ND_, ENCP, NPn_, MU_A, MU_B, MU_S, MU_G, MU_G2, MU_GD, MU_T, MU_AT, P2);
    run_mutual(stream, ma_d, ENCP, NPn_, ENCD, ND_, MU_A, MU_B, MU_S, MU_G, MU_G2, MU_GD, MU_T, MU_AT, D2);

    k_rowmax<<<g1((long)B_ * 128, 256), 256, 0, stream>>>(D2, ND_, XC, 256, 0, B_);
    k_rowmax<<<g1((long)B_ * 128, 256), 256, 0, stream>>>(P2, NPn_, XC, 256, 128, B_);
    gemm(stream, XC, 256, 1, 0, mlp.w1, 1024, mlp.b1, M1, 1024, 1, 0, B_, 1024, 256, 1, 1);
    gemm(stream, M1, 1024, 1, 0, mlp.w2, 512, mlp.b2, M2, 512, 1, 0, B_, 512, 1024, 1, 1);
    gemm(stream, M2, 512, 1, 0, mlp.w3, 1, mlp.b3, (float*)d_out, 1, 1, 0, B_, 1, 512, 0, 1);
}
